// AttnMap_90288802496724
// MI455X (gfx1250) — compile-verified, matrix operand order checked by reading, not run on hardware
//
#include <hip/hip_runtime.h>

// Problem constants (from reference): b=8, t=16, h=w=32 (hw=1024), C=256,
// heads g=8, c=32, q=16, qg=128, features F=256.
#define B_  8
#define T_  16
#define HW_ 1024
#define C_  256
#define G_  8
#define CC_ 32
#define Q_  16
#define QG_ 128
#define F_  256

typedef __attribute__((ext_vector_type(16))) __bf16 v16bf;
typedef __attribute__((ext_vector_type(8)))  float  v8f;

__device__ __forceinline__ unsigned short f32_bf16(float f) {
  union { float f; unsigned int u; } c; c.f = f;
  unsigned int u = c.u;
  u += 0x7FFFu + ((u >> 16) & 1u);   // round-to-nearest-even
  return (unsigned short)(u >> 16);
}

// K index held by VGPR v (0..7), lane-half h (0/1), per CDNA5 16-bit A/B
// fragment layout: lanes 0-15 hold K=0..7,16..23; lanes 16-31 hold K=8..15,24..31,
// two consecutive K values packed per dword. Groups of 4 VGPRs are 16B
// contiguous, so each fragment lowers to two b128 loads.
__device__ __forceinline__ int kpat(int v, int h) {
  return ((v & 4) << 2) + (h << 3) + ((v & 3) << 1);
}

// One-time prep: Wt[n][k] = bf16(W[k][n]); n in [0,256), k in [0,128).
// Transposed so stage-2 B-fragment b128 loads are contiguous in K. W is tiny
// (64 KB as bf16) and stays resident in the 192 MB L2 across all 8192 blocks.
__global__ __launch_bounds__(256)
void prep_w_kernel(const float* __restrict__ W, unsigned short* __restrict__ Wt) {
  int idx = blockIdx.x * 256 + threadIdx.x;     // 32768 elements
  int n = idx >> 7;
  int k = idx & 127;
  Wt[n * QG_ + k] = f32_bf16(W[k * F_ + n]);
}

__global__ __launch_bounds__(256)
void attnmap_fused_kernel(const float* __restrict__ dec,
                          const float* __restrict__ enc,
                          const unsigned short* __restrict__ Wt,
                          const float* __restrict__ bias,
                          float* __restrict__ out) {
  __shared__ unsigned short encT[16][C_];        // 8 KB: 16 spatial rows x (g,c)
  __shared__ unsigned short decT[G_][Q_][CC_];   // 8 KB
  __shared__ unsigned short sT[16][QG_];         // 4 KB: scores, K order = qi*8+g

  const int tid  = threadIdx.x;
  const int mblk = blockIdx.x;   // 64 blocks of 16 spatial rows
  const int t    = blockIdx.y;
  const int b    = blockIdx.z;

  // ---- cooperative coalesced staging (f32 -> bf16) ----
  const long encBase = (((long)(b * T_ + t) * HW_) + (long)mblk * 16) * C_;
#pragma unroll
  for (int i = tid; i < 16 * C_; i += 256) {
    encT[i >> 8][i & 255] = f32_bf16(enc[encBase + i]);
  }
#pragma unroll
  for (int i = tid; i < G_ * Q_ * CC_; i += 256) {
    int g  = i >> 9;
    int qi = (i >> 5) & 15;
    int c  = i & 31;
    decT[g][qi][c] = f32_bf16(dec[(((long)(b * Q_ + qi) * T_) + t) * C_ + g * CC_ + c]);
  }
  __syncthreads();

  const int lane = tid & 31;
  const int wave = tid >> 5;     // 8 waves; wave == head g in stage 1
  const int h    = lane >> 4;    // lane half selects K sub-range
  const int r16  = lane & 15;    // M row (A) / N col (B)

  // ---- Stage 1: S_g[16x16] = Enc_g[16x32] x Dec_g^T[32x16], one WMMA/wave ----
  {
    union { v16bf v; unsigned int u[8]; } A, Bm;
#pragma unroll
    for (int v = 0; v < 8; ++v) {
      int k = kpat(v, h);
      A.u[v]  = *(const unsigned int*)&encT[r16][wave * CC_ + k];
      Bm.u[v] = *(const unsigned int*)&decT[wave][r16][k];
    }
    v8f acc = {};
    acc = __builtin_amdgcn_wmma_f32_16x16x32_bf16(
        /*neg_a=*/false, A.v, /*neg_b=*/false, Bm.v,
        /*c_mod=*/(short)0, acc, /*reuse_a=*/false, /*reuse_b=*/false);
    // C/D layout: VGPR r holds M = r + h*8, N = lane&15. Scatter to LDS with
    // the stage-2 K ordering K = qi*G + g (reference flattens (...,q,g)).
#pragma unroll
    for (int r = 0; r < 8; ++r) {
      int m = r + h * 8;
      sT[m][r16 * G_ + wave] = f32_bf16(acc[r]);
    }
  }
  __syncthreads();

  // ---- Stage 2: out[16x256] = relu(S[16x128] @ W[128x256] + bias) ----
  // 16 N-tiles over 8 waves; both tiles per wave are unconditional so the
  // compiler emits straight-line WMMA code (no exec masking / branches).
  const long outBase = (((long)(b * T_ + t) * HW_) + (long)mblk * 16) * F_;

  auto do_ntile = [&](int nt) {
    v8f acc = {};
    const unsigned short* wrow = Wt + (nt * 16 + r16) * QG_;
#pragma unroll
    for (int kc = 0; kc < 4; ++kc) {             // K = 128 in chunks of 32
      union { v16bf v; unsigned int u[8]; } A, Bm;
#pragma unroll
      for (int v = 0; v < 8; ++v) {
        int k = kc * 32 + kpat(v, h);
        A.u[v]  = *(const unsigned int*)&sT[r16][k];
        Bm.u[v] = *(const unsigned int*)(wrow + k);
      }
      acc = __builtin_amdgcn_wmma_f32_16x16x32_bf16(
          false, A.v, false, Bm.v, (short)0, acc, false, false);
    }
    int   n  = nt * 16 + r16;
    float bn = bias[n];
#pragma unroll
    for (int r = 0; r < 8; ++r) {
      int   m   = r + h * 8;
      float val = acc[r] + bn;
      out[outBase + (long)m * F_ + n] = val > 0.f ? val : 0.f;
    }
  };

  do_ntile(wave);        // tiles 0..7
  do_ntile(wave + 8);    // tiles 8..15

}

extern "C" void kernel_launch(void* const* d_in, const int* in_sizes, int n_in,
                              void* d_out, int out_size, void* d_ws, size_t ws_size,
                              hipStream_t stream) {
  (void)in_sizes; (void)n_in; (void)out_size; (void)ws_size;
  const float* dec  = (const float*)d_in[0];   // btn_dec  [8,256,256]
  const float* enc  = (const float*)d_in[1];   // btn_enc  [8,16,32,32,256]
  const float* W    = (const float*)d_in[2];   // W        [128,256]
  const float* bias = (const float*)d_in[3];   // bias     [256]
  float* out = (float*)d_out;
  unsigned short* Wt = (unsigned short*)d_ws;  // 64 KB bf16 W^T

  prep_w_kernel<<<128, 256, 0, stream>>>(W, Wt);

  dim3 grid(HW_ / 16, T_, B_);                 // (64, 16, 8) = 8192 WGs
  attnmap_fused_kernel<<<grid, 256, 0, stream>>>(dec, enc, Wt, bias, out);
}
